// Decoder_71373766525413
// MI455X (gfx1250) — matrix operand order repaired and checked by reading, not compiled, not run on hardware
//
#include <hip/hip_runtime.h>
#include <hip/hip_bf16.h>

typedef __attribute__((ext_vector_type(16))) __bf16 v16bf;
typedef __attribute__((ext_vector_type(8)))  __bf16 v8bf;
typedef __attribute__((ext_vector_type(8)))  float  v8f;

#define D_FEAT 128
#define KTOT   256   // 2 * D_FEAT (concat width)
#define NKCH   8     // 256 / 32 K-chunks
#define NTIL   8     // 128 / 16 N-tiles (hidden dim)

// ---------- prep 1: cast node features f32 -> bf16 (L2-resident table) ----------
__global__ void cvt_nodes_kernel(const float* __restrict__ x,
                                 __bf16* __restrict__ y, int n) {
    int i = blockIdx.x * blockDim.x + threadIdx.x;
    if (i < n) y[i] = (__bf16)x[i];
}

// ---------- prep 2: pack W1 [128 x 256] into CDNA5 WMMA B-fragment layout ------
// 64 fragments (j0-tile 0..7, kk-chunk 0..7), each 512 bf16:
//   frag[lane*16 + i] = W1[j0*16 + lane%16, kk*32 + (lane>=16 ? 16 : 0) + i]
__global__ void pack_w1_kernel(const float* __restrict__ W1,
                               __bf16* __restrict__ wb) {
    int t = blockIdx.x * blockDim.x + threadIdx.x;   // 0 .. 32767
    int frag = t >> 9, rem = t & 511;
    int lane = rem >> 4, i = rem & 15;
    int j0 = frag >> 3, kk = frag & 7;
    int row = j0 * 16 + (lane & 15);
    int col = kk * 32 + ((lane >> 4) << 4) + i;
    wb[t] = (__bf16)W1[row * KTOT + col];
}

// ---------- main: edge-gather + fused MLP via v_wmma_f32_16x16x32_bf16 --------
// Each wave owns 32 edges (two 16-row A tiles): every B fragment fetched from
// LDS feeds TWO WMMAs, halving LDS bytes per matrix op (the binding resource).
__global__ __launch_bounds__(256)
void edge_mlp_kernel(const __bf16* __restrict__ nodes,
                     const __bf16* __restrict__ w1b,
                     const int* __restrict__ src,
                     const int* __restrict__ dst,
                     const float* __restrict__ b1,
                     const float* __restrict__ w2,
                     const float* __restrict__ b2,
                     float* __restrict__ out_logits,
                     float* __restrict__ out_labels,
                     int nEdges) {
    // W1 B-fragments staged once per workgroup into LDS (64 KB of 320 KB/WGP)
    __shared__ __align__(16) __bf16 ldsB[32768];
    {
        const uint4* g = (const uint4*)w1b;
        uint4* l = (uint4*)ldsB;
        for (int i = threadIdx.x; i < 4096; i += blockDim.x) l[i] = g[i];
    }
    __syncthreads();

    const int lane = threadIdx.x & 31;
    const int n16  = lane & 15;
    const int half = lane >> 4;
    const int wave   = blockIdx.x * (blockDim.x >> 5) + (threadIdx.x >> 5);
    const int nWaves = gridDim.x * (blockDim.x >> 5);
    const int nTiles = (nEdges + 15) >> 4;
    const int nPairs = (nTiles + 1) >> 1;

    // B-fragment load from LDS: per-lane contiguous 32B (two ds_load_b128)
    const __bf16* ldsBase = ldsB + lane * 16;
    auto loadB = [&](int frag) -> v16bf {
        const __bf16* bp = ldsBase + frag * 512;
        v8bf blo = *(const v8bf*)(bp);
        v8bf bhi = *(const v8bf*)(bp + 8);
        return __builtin_shufflevector(blo, bhi,
            0, 1, 2, 3, 4, 5, 6, 7, 8, 9, 10, 11, 12, 13, 14, 15);
    };
    // A fragment gather in contiguous k order (same order as packed B):
    // lane<16: K = kb+{0..15}; lane>=16: kb+{16..31}
    auto loadA = [&](const __bf16* srow, const __bf16* drow, int kk) -> v16bf {
        const __bf16* rp = (kk < 4) ? srow : drow;
        const int kb = (kk & 3) * 32;
        v8bf alo = *(const v8bf*)(rp + kb + (half ? 16 : 0));
        v8bf ahi = *(const v8bf*)(rp + kb + (half ? 24 : 8));
        return __builtin_shufflevector(alo, ahi,
            0, 1, 2, 3, 4, 5, 6, 7, 8, 9, 10, 11, 12, 13, 14, 15);
    };

    // per-lane epilogue constants: column j = t*16 + n16 in the D layout
    float b1f[NTIL], w2f[NTIL];
#pragma unroll
    for (int t = 0; t < NTIL; ++t) {
        b1f[t] = b1[t * 16 + n16];
        w2f[t] = w2[t * 16 + n16];
    }
    const float b2v = b2[0];
    const v8f vzero = {0.f, 0.f, 0.f, 0.f, 0.f, 0.f, 0.f, 0.f};

    // Fused epilogue for one 16-edge tile:
    // D layout: VGPR v, lane l -> row m = v + 8*half, col n = l%16
    auto epilogue = [&](const v8f* acc, int ebase) {
#pragma unroll
        for (int v = 0; v < 8; ++v) {
            float p = 0.f;
#pragma unroll
            for (int t = 0; t < NTIL; ++t) {
                float h = acc[t][v] + b1f[t];
                h = h > 0.f ? h : 0.f;
                p += h * w2f[t];
            }
            // reduce over the 16 columns (masks < 16 stay within each half)
            p += __shfl_xor(p, 8, 32);
            p += __shfl_xor(p, 4, 32);
            p += __shfl_xor(p, 2, 32);
            p += __shfl_xor(p, 1, 32);
            if (n16 == 0) {
                int row = ebase + half * 8 + v;
                if (row < nEdges) out_logits[row] = p + b2v;
            }
        }
    };

    for (int pr = wave; pr < nPairs; pr += nWaves) {
        // Memory clobber: stops LICM from hoisting the 64 loop-invariant LDS
        // B-fragment loads (512 VGPRs!) out of the loop, which would force
        // scratch spills/reloads into the hot loop.
        asm volatile("" ::: "memory");

        const int e0 = pr * 32;
        int eA = e0 + n16;                          // tile 0 rows
        int eB = e0 + 16 + n16;                     // tile 1 rows
        if (eA >= nEdges) eA = nEdges - 1;          // clamp: EXEC stays all-ones
        if (eB >= nEdges) eB = nEdges - 1;
        const __bf16* srowA = nodes + (size_t)src[eA] * D_FEAT;
        const __bf16* drowA = nodes + (size_t)dst[eA] * D_FEAT;
        const __bf16* srowB = nodes + (size_t)src[eB] * D_FEAT;
        const __bf16* drowB = nodes + (size_t)dst[eB] * D_FEAT;

        v8f acc0[NTIL], acc1[NTIL];
#pragma unroll
        for (int t = 0; t < NTIL; ++t) { acc0[t] = vzero; acc1[t] = vzero; }

#pragma unroll
        for (int kk = 0; kk < NKCH; ++kk) {
            v16bf a0 = loadA(srowA, drowA, kk);
            v16bf a1 = loadA(srowB, drowB, kk);
#pragma unroll
            for (int t = 0; t < NTIL; ++t) {
                v16bf b = loadB(t * 8 + kk);
                acc0[t] = __builtin_amdgcn_wmma_f32_16x16x32_bf16(
                    false, a0, false, b, (short)0, acc0[t], false, false);
                acc1[t] = __builtin_amdgcn_wmma_f32_16x16x32_bf16(
                    false, a1, false, b, (short)0, acc1[t], false, false);
            }
        }

        epilogue(acc0, e0);
        epilogue(acc1, e0 + 16);

        int lrow = e0 + lane;                       // 32 labels per pair
        if (lrow < nEdges) out_labels[lrow] = 1.0f;
    }
}

extern "C" void kernel_launch(void* const* d_in, const int* in_sizes, int n_in,
                              void* d_out, int out_size, void* d_ws, size_t ws_size,
                              hipStream_t stream) {
    const float* block_outputs = (const float*)d_in[0];
    const int*   src           = (const int*)d_in[1];
    const int*   dst           = (const int*)d_in[2];
    const float* W1            = (const float*)d_in[3];
    const float* b1            = (const float*)d_in[4];
    const float* W2            = (const float*)d_in[5];
    const float* b2            = (const float*)d_in[6];
    float* out = (float*)d_out;

    const int nNodeElems = in_sizes[0];   // N_NODES * 128
    const int nEdges     = in_sizes[1];

    // workspace: [bf16 node table | packed bf16 W1 fragments]
    __bf16* wsNodes = (__bf16*)d_ws;
    size_t nodesBytes = ((size_t)nNodeElems * sizeof(__bf16) + 255) & ~(size_t)255;
    __bf16* wsB = (__bf16*)((char*)d_ws + nodesBytes);

    cvt_nodes_kernel<<<(nNodeElems + 255) / 256, 256, 0, stream>>>(
        block_outputs, wsNodes, nNodeElems);
    pack_w1_kernel<<<(D_FEAT * KTOT) / 256, 256, 0, stream>>>(W1, wsB);

    edge_mlp_kernel<<<1024, 256, 0, stream>>>(
        wsNodes, wsB, src, dst, b1, W2, b2,
        out /*logits*/, out + nEdges /*labels*/, nEdges);
}
